// STFT_network_3848290697296
// MI455X (gfx1250) — matrix operand order repaired and checked by reading, not compiled, not run on hardware
//
#include <hip/hip_runtime.h>

typedef __attribute__((ext_vector_type(16))) _Float16 v16h;
typedef __attribute__((ext_vector_type(8)))  _Float16 v8h;
typedef __attribute__((ext_vector_type(8)))  float    v8f;

namespace {
constexpr int kWindow   = 2048;
constexpr int kStride   = 512;
constexpr int kFreq     = 1024;
constexpr int kFrames   = 63;
constexpr int kSigLen   = kFrames * kStride;   // 32256
constexpr int kBatch    = 64;
constexpr int kM        = kBatch * kFrames;    // 4032
constexpr int kPadL     = 768;

constexpr int kMTiles   = kM / 16;             // 252
constexpr int kNTiles   = 2048 / 16;           // 128 (real 0..63, imag 64..127)
constexpr int kKTiles   = kWindow / 32;        // 64

// Packed fragment records: per (tile, ktile) -> [hi 1KB][lo 1KB], lane L owns bytes [L*32, L*32+32)
constexpr size_t kAHalves = (size_t)kMTiles * kKTiles * 2 * 512;  // 16,515,072 halves = 33.0 MB
constexpr size_t kBHalves = (size_t)kNTiles * kKTiles * 2 * 512;  // 8,388,608 halves = 16.8 MB
constexpr size_t kWsNeed  = (kAHalves + kBHalves) * sizeof(_Float16); // 49,807,360 B
}

// ---------------------------------------------------------------------------
// Pack A (signal frames) into f16 hi/lo WMMA A-fragments (16x32, wave32 layout):
// lane L (0-15): row M=L, K = koff+e (e<8) / 16+koff+e-8 ; lanes 16-31: koff=8.
// ---------------------------------------------------------------------------
__global__ __launch_bounds__(256)
void pack_a(const float* __restrict__ sig, _Float16* __restrict__ aws)
{
    const int lane = threadIdx.x & 31;
    const int wid  = blockIdx.x * 8 + (threadIdx.x >> 5);   // 0..16127 = mt*64 + kt
    const int m    = (wid >> 6) * 16 + (lane & 15);
    const int kt   = wid & 63;
    const int b    = m / kFrames;
    const int fr   = m - b * kFrames;
    const int kbase = fr * kStride - kPadL + kt * 32;       // signal pos of this tile's K=0
    const float* __restrict__ sp = sig + (size_t)b * kSigLen + kbase;
    const int koff = (lane & 16) ? 8 : 0;

    v16h hi, lo;
    #pragma unroll
    for (int e = 0; e < 16; ++e) {
        const int kl  = (e < 8) ? (koff + e) : (16 + koff + (e - 8));
        const int pos = kbase + kl;
        float x = 0.0f;
        if (pos >= 0 && pos < kSigLen) x = sp[kl];
        const _Float16 h = (_Float16)x;
        hi[e] = h;
        lo[e] = (_Float16)(x - (float)h);
    }
    _Float16* __restrict__ dst = aws + (size_t)wid * 1024 + lane * 16;
    *(v16h*)dst         = hi;
    *(v16h*)(dst + 512) = lo;
}

// ---------------------------------------------------------------------------
// Pack B (DFT kernels) into f16 hi/lo WMMA B-fragments (32x16):
// lane L (0-15): col N=L, K = koff+e (e<8) / 16+koff+e-8 ; lanes 16-31: koff=8.
// ---------------------------------------------------------------------------
__global__ __launch_bounds__(256)
void pack_b(const float* __restrict__ kre, const float* __restrict__ kim,
            _Float16* __restrict__ bws)
{
    const int lane = threadIdx.x & 31;
    const int wid  = blockIdx.x * 8 + (threadIdx.x >> 5);   // 0..8191 = nt*64 + kt
    const int nt   = wid >> 6;
    const int kt   = wid & 63;
    const int col  = nt * 16 + (lane & 15);
    const float* __restrict__ kmat = (col >= kFreq) ? kim : kre;
    const int c  = col & (kFreq - 1);
    const int k0 = kt * 32 + ((lane & 16) ? 8 : 0);

    v16h hi, lo;
    #pragma unroll
    for (int e = 0; e < 16; ++e) {
        const float x = kmat[(size_t)(k0 + e + (e < 8 ? 0 : 8)) * kFreq + c];
        const _Float16 h = (_Float16)x;
        hi[e] = h;
        lo[e] = (_Float16)(x - (float)h);
    }
    _Float16* __restrict__ dst = bws + (size_t)wid * 1024 + lane * 16;
    *(v16h*)dst         = hi;
    *(v16h*)(dst + 512) = lo;
}

// ---------------------------------------------------------------------------
// Main GEMM: pure fragment loads + WMMA. Wave tile 32x32 (4 accumulators,
// 4 independent WMMA chains), block = 8 waves = 64(M) x 128(N), grid (63,16).
// Per k-step: 16 global_load_b128 + 12 v_wmma.
// ---------------------------------------------------------------------------
__global__ __launch_bounds__(256)
void stft_gemm(const _Float16* __restrict__ aws,
               const _Float16* __restrict__ bws,
               float* __restrict__ out)
{
    const int lane = threadIdx.x & 31;
    const int wave = threadIdx.x >> 5;
    const int mt0  = (blockIdx.x * 2 + (wave >> 2)) * 2;    // 0..250, step 2
    const int nt0  = (blockIdx.y * 4 + (wave & 3)) * 2;     // 0..126, step 2

    const _Float16* __restrict__ pa = aws + (size_t)mt0 * kKTiles * 1024 + lane * 16;
    const _Float16* __restrict__ pb = bws + (size_t)nt0 * kKTiles * 1024 + lane * 16;
    constexpr size_t kTileStride = (size_t)kKTiles * 1024;  // next m/n tile (halves)

    v8f acc00 = {}, acc01 = {}, acc10 = {}, acc11 = {};

    #pragma unroll 2
    for (int kt = 0; kt < kKTiles; ++kt) {
        const _Float16* a0 = pa + (size_t)kt * 1024;
        const _Float16* b0 = pb + (size_t)kt * 1024;

        const v16h a0h = *(const v16h*)(a0);
        const v16h a0l = *(const v16h*)(a0 + 512);
        const v16h a1h = *(const v16h*)(a0 + kTileStride);
        const v16h a1l = *(const v16h*)(a0 + kTileStride + 512);
        const v16h b0h = *(const v16h*)(b0);
        const v16h b0l = *(const v16h*)(b0 + 512);
        const v16h b1h = *(const v16h*)(b0 + kTileStride);
        const v16h b1l = *(const v16h*)(b0 + kTileStride + 512);

        acc00 = __builtin_amdgcn_wmma_f32_16x16x32_f16(false, a0h, false, b0h, (short)0, acc00, false, false);
        acc01 = __builtin_amdgcn_wmma_f32_16x16x32_f16(false, a0h, false, b1h, (short)0, acc01, false, false);
        acc10 = __builtin_amdgcn_wmma_f32_16x16x32_f16(false, a1h, false, b0h, (short)0, acc10, false, false);
        acc11 = __builtin_amdgcn_wmma_f32_16x16x32_f16(false, a1h, false, b1h, (short)0, acc11, false, false);

        acc00 = __builtin_amdgcn_wmma_f32_16x16x32_f16(false, a0h, false, b0l, (short)0, acc00, false, false);
        acc01 = __builtin_amdgcn_wmma_f32_16x16x32_f16(false, a0h, false, b1l, (short)0, acc01, false, false);
        acc10 = __builtin_amdgcn_wmma_f32_16x16x32_f16(false, a1h, false, b0l, (short)0, acc10, false, false);
        acc11 = __builtin_amdgcn_wmma_f32_16x16x32_f16(false, a1h, false, b1l, (short)0, acc11, false, false);

        acc00 = __builtin_amdgcn_wmma_f32_16x16x32_f16(false, a0l, false, b0h, (short)0, acc00, false, false);
        acc01 = __builtin_amdgcn_wmma_f32_16x16x32_f16(false, a0l, false, b1h, (short)0, acc01, false, false);
        acc10 = __builtin_amdgcn_wmma_f32_16x16x32_f16(false, a1l, false, b0h, (short)0, acc10, false, false);
        acc11 = __builtin_amdgcn_wmma_f32_16x16x32_f16(false, a1l, false, b1h, (short)0, acc11, false, false);
    }

    // Store: VGPR g -> M = g (lanes 0-15) / 8+g (lanes 16-31); lane&15 -> N column.
    const int  mrow   = (lane & 16) ? 8 : 0;
    const int  fcol   = (nt0 & 63) * 16 + (lane & 15);
    const bool isImag = (nt0 >= 64);
    float* __restrict__ ob = out + (isImag ? (size_t)kM * kFreq : (size_t)0);

    #pragma unroll
    for (int g = 0; g < 8; ++g) {
        const size_t r0 = (size_t)(mt0 * 16 + mrow + g) * kFreq;
        const size_t r1 = r0 + (size_t)16 * kFreq;
        ob[r0 + fcol]      = acc00[g];
        ob[r0 + fcol + 16] = acc01[g];
        ob[r1 + fcol]      = acc10[g];
        ob[r1 + fcol + 16] = acc11[g];
    }
}

// ---------------------------------------------------------------------------
// Fallback (round-1 fused kernel) if workspace is too small for packing.
// ---------------------------------------------------------------------------
namespace {
constexpr int kChunk    = 256;
constexpr int kLdsPitch = kChunk + 8;
}

__global__ __launch_bounds__(256)
void stft_wmma_fused(const float* __restrict__ sig,
                     const float* __restrict__ kre,
                     const float* __restrict__ kim,
                     float* __restrict__ out)
{
    __shared__ __align__(16) _Float16 sAhi[16][kLdsPitch];
    __shared__ __align__(16) _Float16 sAlo[16][kLdsPitch];

    const int tid   = threadIdx.x;
    const int lane  = tid & 31;
    const int wave  = tid >> 5;
    const int strip = blockIdx.x;
    const int ncol0 = blockIdx.y * 128 + wave * 16;
    const bool isImag = (ncol0 >= kFreq);
    const float* __restrict__ kmat = isImag ? kim : kre;
    const int fcol = (ncol0 & (kFreq - 1)) + (lane & 15);

    const int rowA  = lane & 15;
    const int koffA = (lane & 16) ? 8  : 0;
    const int koffB = (lane & 16) ? 8 : 0;
    const float* __restrict__ bcolp = kmat + fcol;

    v8f acc = {};

    for (int ch = 0; ch < kWindow / kChunk; ++ch) {
        const int k0 = ch * kChunk;
        {
            const int k = k0 + tid;
            #pragma unroll
            for (int r = 0; r < 16; ++r) {
                const int m   = strip * 16 + r;
                const int b   = m / kFrames;
                const int fr  = m - b * kFrames;
                const int pos = fr * kStride - kPadL + k;
                float x = 0.0f;
                if (pos >= 0 && pos < kSigLen) x = sig[b * kSigLen + pos];
                const _Float16 h = (_Float16)x;
                sAhi[r][tid] = h;
                sAlo[r][tid] = (_Float16)(x - (float)h);
            }
        }
        __syncthreads();

        #pragma unroll
        for (int s = 0; s < kChunk / 32; ++s) {
            const int kl = s * 32;
            v8h ah0 = *(const v8h*)&sAhi[rowA][kl + koffA];
            v8h ah1 = *(const v8h*)&sAhi[rowA][kl + 16 + koffA];
            v8h al0 = *(const v8h*)&sAlo[rowA][kl + koffA];
            v8h al1 = *(const v8h*)&sAlo[rowA][kl + 16 + koffA];
            v16h ahi = __builtin_shufflevector(ah0, ah1, 0,1,2,3,4,5,6,7,8,9,10,11,12,13,14,15);
            v16h alo = __builtin_shufflevector(al0, al1, 0,1,2,3,4,5,6,7,8,9,10,11,12,13,14,15);

            const float* __restrict__ bp = bcolp + (size_t)(k0 + kl + koffB) * kFreq;
            v16h bhi, blo;
            #pragma unroll
            for (int v = 0; v < 8; ++v) {
                const float b0 = bp[(2 * v     + (v >= 4 ? 8 : 0)) * kFreq];
                const float b1 = bp[(2 * v + 1 + (v >= 4 ? 8 : 0)) * kFreq];
                const _Float16 h0 = (_Float16)b0;
                const _Float16 h1 = (_Float16)b1;
                bhi[2 * v]     = h0;
                bhi[2 * v + 1] = h1;
                blo[2 * v]     = (_Float16)(b0 - (float)h0);
                blo[2 * v + 1] = (_Float16)(b1 - (float)h1);
            }

            acc = __builtin_amdgcn_wmma_f32_16x16x32_f16(false, ahi, false, bhi, (short)0, acc, false, false);
            acc = __builtin_amdgcn_wmma_f32_16x16x32_f16(false, ahi, false, blo, (short)0, acc, false, false);
            acc = __builtin_amdgcn_wmma_f32_16x16x32_f16(false, alo, false, bhi, (short)0, acc, false, false);
        }
        __syncthreads();
    }

    const int mbase = strip * 16 + ((lane & 16) ? 8 : 0);
    float* __restrict__ op = out + (isImag ? (size_t)kM * kFreq : (size_t)0)
                           + (size_t)mbase * kFreq + fcol;
    #pragma unroll
    for (int g = 0; g < 8; ++g) op[(size_t)g * kFreq] = acc[g];
}

// ---------------------------------------------------------------------------
extern "C" void kernel_launch(void* const* d_in, const int* in_sizes, int n_in,
                              void* d_out, int out_size, void* d_ws, size_t ws_size,
                              hipStream_t stream) {
    const float* sig = (const float*)d_in[0];   // [64, 32256, 1] fp32
    const float* kre = (const float*)d_in[1];   // [2048, 1, 1024] fp32
    const float* kim = (const float*)d_in[2];   // [2048, 1, 1024] fp32
    float* out = (float*)d_out;                 // real[4032,1024] ++ imag[4032,1024]

    if (ws_size >= kWsNeed) {
        _Float16* aws = (_Float16*)d_ws;
        _Float16* bws = aws + kAHalves;
        pack_a<<<dim3(kMTiles * kKTiles / 8), dim3(256), 0, stream>>>(sig, aws);
        pack_b<<<dim3(kNTiles * kKTiles / 8), dim3(256), 0, stream>>>(kre, kim, bws);
        stft_gemm<<<dim3(kMTiles / 4, kNTiles / 8), dim3(256), 0, stream>>>(aws, bws, out);
    } else {
        stft_wmma_fused<<<dim3(kMTiles, 16), dim3(256), 0, stream>>>(sig, kre, kim, out);
    }
}
